// ChatSARAN_MLP_3513283248646
// MI455X (gfx1250) — matrix operand order repaired and checked by reading, not compiled, not run on hardware
//
#include <hip/hip_runtime.h>
#include <math.h>
#include <stdint.h>

// ---------------------------------------------------------------------------
// Types for CDNA5 WMMA (wave32): V_WMMA_F32_16X16X32_BF16
// ---------------------------------------------------------------------------
typedef __attribute__((ext_vector_type(16))) __bf16 v16bf;
typedef __attribute__((ext_vector_type(8)))  float  v8f;
typedef __attribute__((ext_vector_type(4)))  unsigned int u32x4;

#define GBM 128
#define GBN 128
#define LDSTRIDE 40   // bf16 elems per LDS row: 32 data + 8 pad (80B, 16B aligned)

enum GemmMode { M_BF16 = 0, M_VTRANS = 1, M_SCORES = 2, M_ADDX = 3,
                M_BIASBF16 = 4, M_MLP2 = 5, M_F32 = 6 };

__device__ __forceinline__ void saran_async_b128(unsigned lds_byte_off,
                                                 const void* gaddr) {
  // GLOBAL_LOAD_ASYNC_TO_LDS_B128: VDST = per-lane LDS byte address,
  // VADDR = per-lane 64-bit global address, SADDR = off. Tracked by ASYNCcnt.
  asm volatile("global_load_async_to_lds_b128 %0, %1, off"
               :: "v"(lds_byte_off), "v"(gaddr)
               : "memory");
}
#define SARAN_WAIT_ASYNC(n) asm volatile("s_wait_asynccnt %0" :: "n"(n) : "memory")

// Generic bf16 WMMA GEMM:  C[M,N] (+epilogue) = A[M,K] * Bt[N,K]^T
// lda = ldb = K for every use in this pipeline; KT = K/32 must be a multiple
// of 4 and >= 4 (all callers use K in {1024, 2048, 4096}).
// Grid: (N/128, M/128, nbatch).
template <int MODE>
__global__ void __launch_bounds__(256)
saran_gemm_wmma(const __bf16* __restrict__ A, const __bf16* __restrict__ Bt,
                long long sAz, long long sBz, long long sCz, int K,
                void* __restrict__ Cptr, int ldc,
                const float* __restrict__ bias,
                float* __restrict__ xf32, __bf16* __restrict__ xbout)
{
  // 4-deep ring of (A,B) tile pairs: 4 * 20KB = 80KB LDS
  __shared__ __align__(16) __bf16 smem[4][2][GBM * LDSTRIDE];

  const int bn = blockIdx.x, bm = blockIdx.y, bz = blockIdx.z;
  if (MODE == M_SCORES) {           // causal: block needs any col <= max row
    if (bn > bm) return;            // uniform exit, EXEC stays full for WMMA
  }

  const __bf16* Ab = A  + (long long)bz * sAz + (long long)(bm * GBM) * K;
  const __bf16* Bb = Bt + (long long)bz * sBz + (long long)(bn * GBN) * K;

  const int tid = threadIdx.x;
  // global->LDS mapping: 512 16B segments per 128x32 tile, 2 per thread
  const int r0 = tid >> 2;               // rows 0..63
  const int k0 = (tid & 3) * 8;          // k offset in elements
  const int r1 = r0 + 64;                // rows 64..127

  const int wid = tid >> 5, lane = tid & 31;
  const int wm = wid >> 1, wn = wid & 1; // 4x2 wave grid
  const int lh = lane >> 4, l16 = lane & 15;

  v8f zacc = {};
  v8f acc[2][4];
#pragma unroll
  for (int i = 0; i < 2; ++i)
#pragma unroll
    for (int j = 0; j < 4; ++j) acc[i][j] = zacc;

  const int KT = K >> 5;                 // multiple of 4, >= 4
  const long long rkA0 = (long long)r0 * K + k0;
  const long long rkA1 = (long long)r1 * K + k0;

  // LDS byte offsets of this thread's two 16B segments in ring buffer 0;
  // other buffers are compile-time multiples of BUFB away.
  const unsigned ldsA0 = (unsigned)(uintptr_t)&smem[0][0][r0 * LDSTRIDE + k0];
  const unsigned ldsA1 = (unsigned)(uintptr_t)&smem[0][0][r1 * LDSTRIDE + k0];
  const unsigned ldsB0 = (unsigned)(uintptr_t)&smem[0][1][r0 * LDSTRIDE + k0];
  const unsigned ldsB1 = (unsigned)(uintptr_t)&smem[0][1][r1 * LDSTRIDE + k0];
  const unsigned BUFB  = (unsigned)sizeof(smem[0]);   // 20480 bytes

  auto issue_tile = [&](int ktn, unsigned bufOff) {
    const __bf16* An = Ab + (ktn << 5);
    const __bf16* Bn = Bb + (ktn << 5);
    saran_async_b128(ldsA0 + bufOff, An + rkA0);
    saran_async_b128(ldsA1 + bufOff, An + rkA1);
    saran_async_b128(ldsB0 + bufOff, Bn + rkA0);
    saran_async_b128(ldsB1 + bufOff, Bn + rkA1);
  };

  auto compute_tile = [&](const __bf16* As, const __bf16* Bs) {
    v16bf afrag[2], bfrag[4];
    // A frag (16x32): lanes 0-15 get K0-7 & K16-23, lanes 16-31 K8-15 & K24-31
#pragma unroll
    for (int i = 0; i < 2; ++i) {
      const __bf16* p = As + (wm * 32 + i * 16 + l16) * LDSTRIDE;
      union { v16bf v; u32x4 q[2]; } u;
      u.q[0] = *(const u32x4*)(p + lh * 8);
      u.q[1] = *(const u32x4*)(p + 16 + lh * 8);
      afrag[i] = u.v;
    }
    // B frag (32x16): lane%16 = column, lanes 0-15 K0-7 & K16-23, lanes 16-31 K8-15 & K24-31
#pragma unroll
    for (int j = 0; j < 4; ++j) {
      const __bf16* p = Bs + (wn * 64 + j * 16 + l16) * LDSTRIDE;
      union { v16bf v; u32x4 q[2]; } u;
      u.q[0] = *(const u32x4*)(p + lh * 8);
      u.q[1] = *(const u32x4*)(p + 16 + lh * 8);
      bfrag[j] = u.v;
    }
#pragma unroll
    for (int i = 0; i < 2; ++i)
#pragma unroll
      for (int j = 0; j < 4; ++j)
        acc[i][j] = __builtin_amdgcn_wmma_f32_16x16x32_bf16(
            false, afrag[i], false, bfrag[j], (short)0, acc[i][j], false, false);
  };

  // prologue: prime 3 tiles of the 4-deep ring (KT >= 4 always)
  issue_tile(0, 0 * BUFB);
  issue_tile(1, 1 * BUFB);
  issue_tile(2, 2 * BUFB);

  // Steady state, fully branch-free per step: wait own tile (in-order async
  // retirement: <=8 outstanding => tile kt's 4 ops landed), barrier (publishes
  // tile kt AND proves ring slot (B+3)&3 was consumed in the previous step), issue
  // tile kt+3, compute tile kt.
#define SARAN_STEP_MAIN(kt0, B)                                              \
  {                                                                          \
    SARAN_WAIT_ASYNC(8);                                                     \
    __syncthreads();                                                         \
    issue_tile((kt0) + 3, (((B) + 3) & 3) * BUFB);                           \
    compute_tile(smem[B][0], smem[B][1]);                                    \
  }

  int kt = 0;
  for (; kt < KT - 4; kt += 4) {         // tiles [0, KT-4): always full steps
    SARAN_STEP_MAIN(kt + 0, 0);
    SARAN_STEP_MAIN(kt + 1, 1);
    SARAN_STEP_MAIN(kt + 2, 2);
    SARAN_STEP_MAIN(kt + 3, 3);
  }
#undef SARAN_STEP_MAIN

  // drain: tiles KT-4..KT-1 in ring slots 0..3 (KT % 4 == 0)
  SARAN_WAIT_ASYNC(8);
  __syncthreads();
  issue_tile(KT - 1, 3 * BUFB);
  compute_tile(smem[0][0], smem[0][1]);

  SARAN_WAIT_ASYNC(8);
  __syncthreads();
  compute_tile(smem[1][0], smem[1][1]);

  SARAN_WAIT_ASYNC(4);
  __syncthreads();
  compute_tile(smem[2][0], smem[2][1]);

  SARAN_WAIT_ASYNC(0);
  __syncthreads();
  compute_tile(smem[3][0], smem[3][1]);

  // Epilogue. C/D layout: lane%16 = N, VGPR r -> M = 8*(lane/16) + r
  const int orow = bm * GBM + wm * 32 + lh * 8;
  const int ocol = bn * GBN + wn * 64 + l16;
#pragma unroll
  for (int i = 0; i < 2; ++i) {
#pragma unroll
    for (int j = 0; j < 4; ++j) {
#pragma unroll
      for (int r = 0; r < 8; ++r) {
        const int row = orow + i * 16 + r;
        const int col = ocol + j * 16;
        const float v = acc[i][j][r];
        if (MODE == M_BF16) {
          ((__bf16*)Cptr)[(long long)row * ldc + col] = (__bf16)v;
        } else if (MODE == M_VTRANS) {
          // v-projection row = b*2048+s, col = d  ->  vt[b][d][s]
          const long long o = ((long long)(row >> 11) << 21) +
                              ((long long)col << 11) + (long long)(row & 2047);
          ((__bf16*)Cptr)[o] = (__bf16)v;
        } else if (MODE == M_SCORES) {
          ((float*)Cptr)[(long long)bz * sCz + (long long)row * ldc + col] = v;
        } else if (MODE == M_ADDX) {
          float* C = (float*)Cptr + (long long)bz * sCz;
          C[(long long)row * ldc + col] += v;
        } else if (MODE == M_BIASBF16) {
          ((__bf16*)Cptr)[(long long)row * ldc + col] = (__bf16)(v + bias[col]);
        } else if (MODE == M_MLP2) {
          const long long o = (long long)row * ldc + col;
          const float xn = xf32[o] + 0.1f * (v + bias[col]);
          xf32[o] = xn;
          xbout[o] = (__bf16)xn;
        } else { // M_F32
          ((float*)Cptr)[(long long)row * ldc + col] = v;
        }
      }
    }
  }
}

// ---------------------------------------------------------------------------
// Support kernels
// ---------------------------------------------------------------------------
__global__ void saran_convert_bf16(const float* __restrict__ in,
                                   __bf16* __restrict__ out, size_t n) {
  size_t i = (size_t)blockIdx.x * blockDim.x + threadIdx.x;
  const size_t stride = (size_t)gridDim.x * blockDim.x;
  for (; i < n; i += stride) out[i] = (__bf16)in[i];
}

// in[R][C] f32 -> out[C][R] bf16
__global__ void __launch_bounds__(256)
saran_tconvert_bf16(const float* __restrict__ in, __bf16* __restrict__ out,
                    int R, int C) {
  __shared__ float tile[32][33];
  const int c0 = blockIdx.x * 32, r0 = blockIdx.y * 32;
  const int tx = threadIdx.x & 31, ty = threadIdx.x >> 5;
  for (int i = ty; i < 32; i += 8)
    tile[i][tx] = in[(size_t)(r0 + i) * C + (c0 + tx)];
  __syncthreads();
  for (int i = ty; i < 32; i += 8)
    out[(size_t)(c0 + i) * R + (r0 + tx)] = (__bf16)tile[tx][i];
}

// x = tok_emb[idx] + pos_emb[t];  write fp32 residual + bf16 copy
__global__ void saran_embed(const int* __restrict__ idx,
                            const float* __restrict__ tok,
                            const float* __restrict__ pos,
                            float* __restrict__ xf, __bf16* __restrict__ xb) {
  const int g = blockIdx.x;                 // 0..4095 (b*2048+t)
  const int t = g & 2047;
  const int token = idx[g];
  const float* te = tok + (size_t)token * 1024;
  const float* pe = pos + (size_t)t * 1024;
  float* xo = xf + (size_t)g * 1024;
  __bf16* xbo = xb + (size_t)g * 1024;
  for (int d = threadIdx.x; d < 1024; d += blockDim.x) {
    const float v = te[d] + pe[d];
    xo[d] = v;
    xbo[d] = (__bf16)v;
  }
}

// one block per row: causal softmax over scores[g][0..t], write bf16, zero tail
__global__ void __launch_bounds__(256)
saran_softmax(const float* __restrict__ scores, __bf16* __restrict__ attn,
              float scale) {
  const int g = blockIdx.x;
  const int t = g & 2047;
  const int n = t + 1;
  const float* srow = scores + ((long long)g << 11);
  __bf16* arow = attn + ((long long)g << 11);
  __shared__ float red[8];
  const int lane = threadIdx.x & 31, wid = threadIdx.x >> 5;

  float lm = -3.4e38f;
  for (int i = threadIdx.x; i < n; i += 256) lm = fmaxf(lm, srow[i]);
  for (int o = 16; o; o >>= 1) lm = fmaxf(lm, __shfl_xor(lm, o, 32));
  if (!lane) red[wid] = lm;
  __syncthreads();
  float m = fmaxf(fmaxf(fmaxf(red[0], red[1]), fmaxf(red[2], red[3])),
                  fmaxf(fmaxf(red[4], red[5]), fmaxf(red[6], red[7])));
  m *= scale;

  float ls = 0.f;
  for (int i = threadIdx.x; i < n; i += 256) ls += __expf(srow[i] * scale - m);
  for (int o = 16; o; o >>= 1) ls += __shfl_xor(ls, o, 32);
  __syncthreads();
  if (!lane) red[wid] = ls;
  __syncthreads();
  const float s = red[0] + red[1] + red[2] + red[3] +
                  red[4] + red[5] + red[6] + red[7];
  const float inv = 1.f / s;
  for (int i = threadIdx.x; i < 2048; i += 256)
    arow[i] = (i < n) ? (__bf16)(__expf(srow[i] * scale - m) * inv)
                      : (__bf16)0.0f;
}

// GEGLU: out[t][j] = h[t][j] * gelu_exact(h[t][j+4096]) ; h is [4096][8192] bf16
__global__ void saran_geglu(const __bf16* __restrict__ h,
                            __bf16* __restrict__ out) {
  size_t i = (size_t)blockIdx.x * blockDim.x + threadIdx.x;
  const size_t stride = (size_t)gridDim.x * blockDim.x;
  const size_t total = (size_t)4096 * 4096;
  for (; i < total; i += stride) {
    const size_t t = i >> 12, j = i & 4095;
    const float a = (float)h[(t << 13) + j];
    const float g = (float)h[(t << 13) + j + 4096];
    const float gelu = 0.5f * g * (1.0f + erff(g * 0.70710678118654752f));
    out[i] = (__bf16)(a * gelu);
  }
}

// ---------------------------------------------------------------------------
// Host orchestration
// ---------------------------------------------------------------------------
static char* ws_take(char** p, size_t bytes) {
  char* r = *p;
  *p += (bytes + 255) & ~(size_t)255;
  return r;
}

extern "C" void kernel_launch(void* const* d_in, const int* in_sizes, int n_in,
                              void* d_out, int out_size, void* d_ws, size_t ws_size,
                              hipStream_t stream) {
  (void)in_sizes; (void)n_in; (void)out_size; (void)ws_size;
  const int*   idx = (const int*)d_in[0];
  const float* tok = (const float*)d_in[1];
  const float* pos = (const float*)d_in[2];
  const float* Wq  = (const float*)d_in[3];
  const float* Wk  = (const float*)d_in[4];
  const float* Wv  = (const float*)d_in[5];
  const float* W1  = (const float*)d_in[6];
  const float* b1  = (const float*)d_in[7];
  const float* W2  = (const float*)d_in[8];
  const float* b2  = (const float*)d_in[9];
  float* logits = (float*)d_out;

  char* p = (char*)d_ws;
  float*  xf32  = (float*) ws_take(&p, (size_t)4096 * 1024 * 4);
  __bf16* xb    = (__bf16*)ws_take(&p, (size_t)4096 * 1024 * 2);
  __bf16* tokb  = (__bf16*)ws_take(&p, (size_t)32000 * 1024 * 2);
  __bf16* WqT   = (__bf16*)ws_take(&p, (size_t)1024 * 1024 * 2);
  __bf16* WkT   = (__bf16*)ws_take(&p, (size_t)1024 * 1024 * 2);
  __bf16* WvT   = (__bf16*)ws_take(&p, (size_t)1024 * 1024 * 2);
  __bf16* W1T   = (__bf16*)ws_take(&p, (size_t)8192 * 1024 * 2);
  __bf16* W2T   = (__bf16*)ws_take(&p, (size_t)1024 * 4096 * 2);
  __bf16* qb    = (__bf16*)ws_take(&p, (size_t)4096 * 1024 * 2);
  __bf16* kb    = (__bf16*)ws_take(&p, (size_t)4096 * 1024 * 2);
  __bf16* vt    = (__bf16*)ws_take(&p, (size_t)2 * 1024 * 2048 * 2);
  float*  sc    = (float*) ws_take(&p, (size_t)2 * 2048 * 2048 * 4);
  __bf16* attn  = (__bf16*)ws_take(&p, (size_t)2 * 2048 * 2048 * 2);
  __bf16* h     = (__bf16*)ws_take(&p, (size_t)4096 * 8192 * 2);
  __bf16* mlpin = (__bf16*)ws_take(&p, (size_t)4096 * 4096 * 2);

  // --- weight / embedding precision prep (bf16; tokb also serves big GEMM) ---
  saran_convert_bf16<<<4096, 256, 0, stream>>>(tok, tokb, (size_t)32000 * 1024);
  saran_tconvert_bf16<<<dim3(32, 32), 256, 0, stream>>>(Wq, WqT, 1024, 1024);
  saran_tconvert_bf16<<<dim3(32, 32), 256, 0, stream>>>(Wk, WkT, 1024, 1024);
  saran_tconvert_bf16<<<dim3(32, 32), 256, 0, stream>>>(Wv, WvT, 1024, 1024);
  saran_tconvert_bf16<<<dim3(256, 32), 256, 0, stream>>>(W1, W1T, 1024, 8192);
  saran_tconvert_bf16<<<dim3(32, 128), 256, 0, stream>>>(W2, W2T, 4096, 1024);

  // --- x = tok_emb[idx] + pos_emb ---
  saran_embed<<<4096, 256, 0, stream>>>(idx, tok, pos, xf32, xb);

  // --- QKV projections (M=4096,N=1024,K=1024) ---
  saran_gemm_wmma<M_BF16><<<dim3(8, 32, 1), 256, 0, stream>>>(
      xb, WqT, 0, 0, 0, 1024, qb, 1024, nullptr, nullptr, nullptr);
  saran_gemm_wmma<M_BF16><<<dim3(8, 32, 1), 256, 0, stream>>>(
      xb, WkT, 0, 0, 0, 1024, kb, 1024, nullptr, nullptr, nullptr);
  saran_gemm_wmma<M_VTRANS><<<dim3(8, 32, 1), 256, 0, stream>>>(
      xb, WvT, 0, 0, 0, 1024, vt, 0, nullptr, nullptr, nullptr);

  // --- scores = q @ k^T (batched, causal block skip) ---
  saran_gemm_wmma<M_SCORES><<<dim3(16, 16, 2), 256, 0, stream>>>(
      qb, kb, (long long)2048 * 1024, (long long)2048 * 1024,
      (long long)2048 * 2048, 1024, sc, 2048, nullptr, nullptr, nullptr);

  // --- causal softmax (scale = 1/sqrt(1024) = 1/32) ---
  saran_softmax<<<4096, 256, 0, stream>>>(sc, attn, 0.03125f);

  // --- x += attn @ v  (batched; Bt = v^T) ---
  saran_gemm_wmma<M_ADDX><<<dim3(8, 16, 2), 256, 0, stream>>>(
      attn, vt, (long long)2048 * 2048, (long long)1024 * 2048,
      (long long)2048 * 1024, 2048, xf32, 1024, nullptr, nullptr, nullptr);

  // --- refresh bf16 activations after attention residual ---
  saran_convert_bf16<<<2048, 256, 0, stream>>>(xf32, xb, (size_t)4096 * 1024);

  // --- h = x @ W1 + b1  (M=4096,N=8192,K=1024) ---
  saran_gemm_wmma<M_BIASBF16><<<dim3(64, 32, 1), 256, 0, stream>>>(
      xb, W1T, 0, 0, 0, 1024, h, 8192, b1, nullptr, nullptr);

  // --- GEGLU ---
  saran_geglu<<<8192, 256, 0, stream>>>(h, mlpin);

  // --- x += 0.1*(mlpin @ W2 + b2); emit fp32 + bf16 x ---
  saran_gemm_wmma<M_MLP2><<<dim3(8, 32, 1), 256, 0, stream>>>(
      mlpin, W2T, 0, 0, 0, 4096, xf32, 1024, b2, xf32, xb);

  // --- logits = x @ tok_emb^T  (M=4096,N=32000,K=1024) ---
  saran_gemm_wmma<M_F32><<<dim3(250, 32, 1), 256, 0, stream>>>(
      xb, tokb, 0, 0, 0, 1024, logits, 32000, nullptr, nullptr, nullptr);
}
